// SpikeNeuralNetwork_34660386079150
// MI455X (gfx1250) — compile-verified, matrix operand order checked by reading, not run on hardware
//
#include <hip/hip_runtime.h>
#include <hip/hip_bf16.h>

// ---------------------------------------------------------------------------
// SNN: out = ( LIF_counts( x @ W_in^T + b_in ) / 20 ) @ W_out^T + b_out
//   GEMM1 in fp32 WMMA (threshold dynamics are chaotic -> need fp32 inp)
//   GEMM2 in bf16 WMMA (counts are exact ints in bf16; W_out split hi+lo)
// ---------------------------------------------------------------------------

typedef __attribute__((ext_vector_type(2)))  float  v2f;
typedef __attribute__((ext_vector_type(8)))  float  v8f;
typedef __attribute__((ext_vector_type(8)))  __bf16 v8bf;
typedef __attribute__((ext_vector_type(16))) __bf16 v16bf;

#define BATCH   4096
#define IN_DIM  1024
#define HID     4096
#define OUT_DIM 1024
#define STEPS   20

// ---------------------------------------------------------------------------
// Kernel 0: split W_out (f32) into bf16 hi/lo planes (w == hi + lo to ~2^-16)
// ---------------------------------------------------------------------------
__global__ __launch_bounds__(256) void snn_wsplit(
    const float* __restrict__ W, __bf16* __restrict__ Whi,
    __bf16* __restrict__ Wlo, int n)
{
    int i = blockIdx.x * blockDim.x + threadIdx.x;
    if (i < n) {
        float  w  = W[i];
        __bf16 hi = (__bf16)w;                 // RNE
        Whi[i] = hi;
        Wlo[i] = (__bf16)(w - (float)hi);
    }
}

// ---------------------------------------------------------------------------
// Kernel 1: fused  inp = x @ W_in^T + b_in  (f32 WMMA 16x16x4)  + 20-step LIF
// Wave tile: 32 (batch rows) x 64 (hidden cols). Block = 8 waves = 64 x 256.
//
// f32 16x16x4 fragment layout (ISA 7.12.2):
//   A (16x4, M x K): lane = m + 16*(k>>1), vgpr = k&1
//     -> lane loads x[b0 + lane%16][k + 2*(lane/16) .. +1]   (one b64)
//   B (4x16, K x N) mirrors A with lanes indexing N; B[k][n] = W_in[n][k]
//     -> lane loads W_in[h0 + lane%16][k + 2*(lane/16) .. +1] (one b64)
//   C/D (16x16): lane holds col n = lane%16, rows m = vgpr + 8*(lane/16)
// ---------------------------------------------------------------------------
__global__ __launch_bounds__(256) void snn_gemm1_lif(
    const float* __restrict__ x, const float* __restrict__ Win,
    const float* __restrict__ b_in, __bf16* __restrict__ counts)
{
    const int lane = threadIdx.x & 31;
    const int wave = threadIdx.x >> 5;
    const int b0 = blockIdx.y * 64  + (wave >> 2) * 32;   // batch base of wave
    const int h0 = blockIdx.x * 256 + (wave & 3) * 64;    // hidden base of wave

    const int lm = lane & 15;          // row/col index within 16
    const int lk = (lane >> 4) * 2;    // k sub-offset (0 or 2)

    v8f acc[2][4];
    #pragma unroll
    for (int i = 0; i < 2; ++i)
        #pragma unroll
        for (int j = 0; j < 4; ++j) acc[i][j] = (v8f)0.0f;

    const float* ap0 = x   + (size_t)(b0      + lm) * IN_DIM + lk;
    const float* ap1 = x   + (size_t)(b0 + 16 + lm) * IN_DIM + lk;
    const float* bp0 = Win + (size_t)(h0      + lm) * IN_DIM + lk;
    const float* bp1 = Win + (size_t)(h0 + 16 + lm) * IN_DIM + lk;
    const float* bp2 = Win + (size_t)(h0 + 32 + lm) * IN_DIM + lk;
    const float* bp3 = Win + (size_t)(h0 + 48 + lm) * IN_DIM + lk;

    #pragma unroll 4
    for (int k = 0; k < IN_DIM; k += 4) {
        v2f a0 = *(const v2f*)(ap0 + k);
        v2f a1 = *(const v2f*)(ap1 + k);
        v2f b0v = *(const v2f*)(bp0 + k);
        v2f b1v = *(const v2f*)(bp1 + k);
        v2f b2v = *(const v2f*)(bp2 + k);
        v2f b3v = *(const v2f*)(bp3 + k);

        acc[0][0] = __builtin_amdgcn_wmma_f32_16x16x4_f32(false, a0, false, b0v, (short)0, acc[0][0], false, false);
        acc[0][1] = __builtin_amdgcn_wmma_f32_16x16x4_f32(false, a0, false, b1v, (short)0, acc[0][1], false, false);
        acc[0][2] = __builtin_amdgcn_wmma_f32_16x16x4_f32(false, a0, false, b2v, (short)0, acc[0][2], false, false);
        acc[0][3] = __builtin_amdgcn_wmma_f32_16x16x4_f32(false, a0, false, b3v, (short)0, acc[0][3], false, false);
        acc[1][0] = __builtin_amdgcn_wmma_f32_16x16x4_f32(false, a1, false, b0v, (short)0, acc[1][0], false, false);
        acc[1][1] = __builtin_amdgcn_wmma_f32_16x16x4_f32(false, a1, false, b1v, (short)0, acc[1][1], false, false);
        acc[1][2] = __builtin_amdgcn_wmma_f32_16x16x4_f32(false, a1, false, b2v, (short)0, acc[1][2], false, false);
        acc[1][3] = __builtin_amdgcn_wmma_f32_16x16x4_f32(false, a1, false, b3v, (short)0, acc[1][3], false, false);
    }

    // ---- LIF dynamics in registers, then store spike counts (exact bf16 ints)
    const int rbase = 8 * (lane >> 4);     // row offset within 16x16 tile
    #pragma unroll
    for (int j = 0; j < 4; ++j) {
        const float bias = b_in[h0 + 16 * j + lm];
        #pragma unroll
        for (int i = 0; i < 2; ++i) {
            #pragma unroll
            for (int r = 0; r < 8; ++r) {
                const float inp = acc[i][j][r] + bias;
                float v = 0.0f, c = 0.0f;
                #pragma unroll
                for (int s = 0; s < STEPS; ++s) {
                    v = v * 0.9f + inp;                       // leak + drive
                    float sp = (v > 1.0f) ? 1.0f : 0.0f;      // threshold
                    c += sp;
                    v *= (1.0f - sp);                         // hard reset
                }
                const size_t row = (size_t)(b0 + 16 * i + rbase + r);
                counts[row * HID + h0 + 16 * j + lm] = (__bf16)c;  // exact
            }
        }
    }
}

// ---------------------------------------------------------------------------
// Kernel 2: out = (counts @ W_out^T) * (1/20) + b_out   (bf16 WMMA 16x16x32)
// Wave tile: 32 (batch) x 64 (out cols). Block = 8 waves = 64 x 256.
//
// bf16 A-fragment 16x32 (ISA 7.12.2): lane holds row m = lane%16,
//   K = 8*(lane/16)+{0..7} in elements 0..7 and K = 16+8*(lane/16)+{0..7}
//   in elements 8..15  -> two b128 loads per fragment.
// B mirrors A with lanes indexing N: B[k][n] = W[n][k] -> same addressing.
// ---------------------------------------------------------------------------
__global__ __launch_bounds__(256) void snn_gemm2(
    const __bf16* __restrict__ counts, const __bf16* __restrict__ Whi,
    const __bf16* __restrict__ Wlo, const float* __restrict__ b_out,
    float* __restrict__ out)
{
    const int lane = threadIdx.x & 31;
    const int wave = threadIdx.x >> 5;
    const int b0 = blockIdx.y * 64  + (wave >> 2) * 32;
    const int o0 = blockIdx.x * 256 + (wave & 3) * 64;

    const int lm  = lane & 15;
    const int lk8 = (lane >> 4) * 8;

    v8f acc[2][4];
    #pragma unroll
    for (int i = 0; i < 2; ++i)
        #pragma unroll
        for (int j = 0; j < 4; ++j) acc[i][j] = (v8f)0.0f;

    const __bf16* arow[2];
    arow[0] = counts + (size_t)(b0      + lm) * HID + lk8;
    arow[1] = counts + (size_t)(b0 + 16 + lm) * HID + lk8;

    for (int k = 0; k < HID; k += 32) {
        union { v16bf v; v8bf h[2]; } a[2], bh[4], bl[4];
        #pragma unroll
        for (int i = 0; i < 2; ++i) {
            a[i].h[0] = *(const v8bf*)(arow[i] + k);
            a[i].h[1] = *(const v8bf*)(arow[i] + k + 16);
        }
        #pragma unroll
        for (int j = 0; j < 4; ++j) {
            const size_t off = (size_t)(o0 + 16 * j + lm) * HID + k + lk8;
            bh[j].h[0] = *(const v8bf*)(Whi + off);
            bh[j].h[1] = *(const v8bf*)(Whi + off + 16);
            bl[j].h[0] = *(const v8bf*)(Wlo + off);
            bl[j].h[1] = *(const v8bf*)(Wlo + off + 16);
        }
        #pragma unroll
        for (int i = 0; i < 2; ++i)
            #pragma unroll
            for (int j = 0; j < 4; ++j) {
                acc[i][j] = __builtin_amdgcn_wmma_f32_16x16x32_bf16(
                    false, a[i].v, false, bh[j].v, (short)0, acc[i][j], false, false);
                acc[i][j] = __builtin_amdgcn_wmma_f32_16x16x32_bf16(
                    false, a[i].v, false, bl[j].v, (short)0, acc[i][j], false, false);
            }
    }

    const int rbase = 8 * (lane >> 4);
    #pragma unroll
    for (int j = 0; j < 4; ++j) {
        const float bo = b_out[o0 + 16 * j + lm];
        #pragma unroll
        for (int i = 0; i < 2; ++i) {
            #pragma unroll
            for (int r = 0; r < 8; ++r) {
                const size_t row = (size_t)(b0 + 16 * i + rbase + r);
                out[row * OUT_DIM + o0 + 16 * j + lm] =
                    acc[i][j][r] * (1.0f / (float)STEPS) + bo;
            }
        }
    }
}

// ---------------------------------------------------------------------------
extern "C" void kernel_launch(void* const* d_in, const int* in_sizes, int n_in,
                              void* d_out, int out_size, void* d_ws, size_t ws_size,
                              hipStream_t stream) {
    const float* x     = (const float*)d_in[0];
    const float* W_in  = (const float*)d_in[1];
    const float* b_in  = (const float*)d_in[2];
    const float* W_out = (const float*)d_in[3];
    const float* b_out = (const float*)d_in[4];
    // d_in[5] = steps (==20, baked into STEPS)
    float* out = (float*)d_out;

    // Workspace: counts bf16 [4096x4096] (32 MB) | Whi bf16 (8 MB) | Wlo bf16 (8 MB)
    __bf16* counts = (__bf16*)d_ws;
    __bf16* Whi    = (__bf16*)((char*)d_ws + (size_t)BATCH * HID * sizeof(__bf16));
    __bf16* Wlo    = Whi + (size_t)OUT_DIM * HID;

    const int nW = OUT_DIM * HID;
    snn_wsplit<<<(nW + 255) / 256, 256, 0, stream>>>(W_out, Whi, Wlo, nW);

    // GEMM1 + LIF: grid covers [BATCH/64] x [HID/256]
    snn_gemm1_lif<<<dim3(HID / 256, BATCH / 64), 256, 0, stream>>>(
        x, W_in, b_in, counts);

    // GEMM2: grid covers [BATCH/64] x [OUT_DIM/256]
    snn_gemm2<<<dim3(OUT_DIM / 256, BATCH / 64), 256, 0, stream>>>(
        counts, Whi, Wlo, b_out, out);
}
